// ConvMapCell_67860483277391
// MI455X (gfx1250) — compile-verified, matrix operand order checked by reading, not run on hardware
//
#include <hip/hip_runtime.h>

// ---------------------------------------------------------------------------
// ConvMapCell forward on MI455X (gfx1250): every 3x3 conv -> im2col GEMM on
// v_wmma_f32_16x16x32_bf16 (bf16 inputs, f32 accumulate), BN/bias/ReLU fused
// into the GEMM epilogue. M = 1024*36 = 36864 for every GEMM.
// Wave tile 32x64 (2 A-frags x 4 B-frags, 8 f32 accumulators) -> 1.5 b128
// loads per WMMA; block = 8 waves tiling 128x128.
// ---------------------------------------------------------------------------

typedef __attribute__((ext_vector_type(16))) __bf16        v16bf;
typedef __attribute__((ext_vector_type(8)))  float         v8f;
typedef __attribute__((ext_vector_type(4)))  unsigned int  v4u;

union FragU { v16bf bf; v4u u[2]; };

#define MTOT    36864   // 1024 samples * 36 pixels
#define MBLOCKS 288     // MTOT / 128

#define FLAG_RELU 1
#define FLAG_BN   2
#define FLAG_ACC  4

__device__ __forceinline__ unsigned short f2bf(float f) {
  union { float f; unsigned u; } x; x.f = f;
  unsigned u = x.u;
  unsigned r = u + 0x7FFFu + ((u >> 16) & 1u);   // round-to-nearest-even
  return (unsigned short)(r >> 16);
}

// --- weight prep: fp32 conv weights -> bf16 (N rows x Kpad cols), K zero-pad.
// mode 0: conv3x3  src (N,Cin,3,3),  Wb[n][c*9+t] = src[n][c][t]
// mode 1: convT3x3 src (Cin,N,3,3),  Wb[n][c*9+t] = src[c][n][8-t]  (flip)
// mode 2: 1x1      src (N,Cin),      Wb[n][k]     = src[n][k]
__global__ void prep_w_kernel(unsigned short* __restrict__ dst,
                              const float* __restrict__ src,
                              int N, int Cin, int Kpad, int mode, long total) {
  long i = (long)blockIdx.x * blockDim.x + threadIdx.x;
  if (i >= total) return;
  int k = (int)(i % Kpad);
  int n = (int)(i / Kpad);
  float v = 0.f;
  if (mode == 2) {
    if (k < Cin) v = src[(size_t)n * Cin + k];
  } else {
    int K9 = Cin * 9;
    if (k < K9) {
      int c = k / 9, t = k - c * 9;
      v = (mode == 0) ? src[((size_t)n * Cin + c) * 9 + t]
                      : src[((size_t)c * N + n) * 9 + (8 - t)];
    }
  }
  dst[i] = f2bf(v);
}

// --- im2col (SAME, 3x3) or layout transform (1x1): fp32 NCHW -> bf16 (M x Kpad)
__global__ void im2col_kernel(unsigned short* __restrict__ dst,
                              const float* __restrict__ src,
                              int C, int Kpad, int ks3, long total) {
  long i = (long)blockIdx.x * blockDim.x + threadIdx.x;
  if (i >= total) return;
  int  k  = (int)(i % Kpad);
  long mr = i / Kpad;
  int  p  = (int)(mr % 36);
  int  b  = (int)(mr / 36);
  float v = 0.f;
  if (ks3) {
    if (k < C * 9) {
      int c = k / 9, t = k - c * 9;
      int yy = p / 6 + t / 3 - 1;
      int xx = p % 6 + t % 3 - 1;
      if ((unsigned)yy < 6u && (unsigned)xx < 6u)
        v = src[((size_t)b * C + c) * 36 + yy * 6 + xx];
    }
  } else {
    if (k < C) v = src[((size_t)b * C + k) * 36 + p];
  }
  dst[i] = f2bf(v);
}

// --- channel concat helper: copy (B,C,36) into (B,Ctot,36) at channel offset
__global__ void copy_ch_kernel(float* __restrict__ dst, const float* __restrict__ src,
                               int Ctot, int choff, int C, long total) {
  long i = (long)blockIdx.x * blockDim.x + threadIdx.x;
  if (i >= total) return;
  int  p   = (int)(i % 36);
  long cb  = i / 36;
  int  c   = (int)(cb % C);
  int  b   = (int)(cb / C);
  dst[((size_t)b * Ctot + choff + c) * 36 + p] = src[i];
}

// --- WMMA GEMM: D(M x N) = A(M x Kpad, bf16) * Bw^T (Bw is N x Kpad, bf16)
//     out written as NCHW fp32: out[(b*N + n)*36 + p], m = b*36 + p.
//     epilogue: +bias, optional BN (g,b,mean,var), optional ReLU, optional +=.
//     Wave tile 32(M) x 64(N); block = 8 waves as 4(M) x 2(N) = 128 x 128.
__global__ __launch_bounds__(256) void gemm_wmma_kernel(
    float* __restrict__ out,
    const unsigned short* __restrict__ A,
    const unsigned short* __restrict__ Bw,
    int N, int Kpad,
    const float* __restrict__ bias,
    const float* __restrict__ bn_g, const float* __restrict__ bn_b,
    const float* __restrict__ bn_m, const float* __restrict__ bn_v,
    int flags) {
  const int lane = threadIdx.x & 31;
  const int wave = threadIdx.x >> 5;
  const int m0 = blockIdx.y * 128 + (wave & 3) * 32;    // 4 waves in M
  const int n0 = blockIdx.x * 128 + (wave >> 2) * 64;   // 2 waves in N
  const int row = lane & 15;
  const int kh  = lane >> 4;                            // K-half select

  // A fragment rows (always in-bounds: M % 128 == 0)
  const unsigned short* pa0 = A + (size_t)(m0 + row) * Kpad + kh * 8;
  const unsigned short* pa1 = pa0 + (size_t)16 * Kpad;
  // B fragment rows = output channels; clamp OOB rows to 0 (outputs guarded)
  const unsigned short* pb[4];
#pragma unroll
  for (int j = 0; j < 4; ++j) {
    int cb = n0 + j * 16 + row;
    pb[j] = Bw + (size_t)(cb < N ? cb : 0) * Kpad + kh * 8;
  }

  v8f acc0[4] = {v8f{}, v8f{}, v8f{}, v8f{}};
  v8f acc1[4] = {v8f{}, v8f{}, v8f{}, v8f{}};

  for (int kk = 0; kk < Kpad; kk += 32) {
    FragU a0, a1, b[4];
    a0.u[0] = *(const v4u*)(pa0 + kk); a0.u[1] = *(const v4u*)(pa0 + kk + 16);
    a1.u[0] = *(const v4u*)(pa1 + kk); a1.u[1] = *(const v4u*)(pa1 + kk + 16);
#pragma unroll
    for (int j = 0; j < 4; ++j) {
      b[j].u[0] = *(const v4u*)(pb[j] + kk);
      b[j].u[1] = *(const v4u*)(pb[j] + kk + 16);
    }
#pragma unroll
    for (int j = 0; j < 4; ++j) {
      acc0[j] = __builtin_amdgcn_wmma_f32_16x16x32_bf16(false, a0.bf, false, b[j].bf,
                                                        (short)0, acc0[j], false, false);
      acc1[j] = __builtin_amdgcn_wmma_f32_16x16x32_bf16(false, a1.bf, false, b[j].bf,
                                                        (short)0, acc1[j], false, false);
    }
  }

  const int nloc = lane & 15;
  auto epilogue = [&](v8f acc, int mb, int nn) {
    if (nn >= N) return;
    float bb    = bias[nn];
    float scale = 1.f, shift = bb;
    if (flags & FLAG_BN) {
      float inv = bn_g[nn] * rsqrtf(bn_v[nn] + 1e-5f);
      scale = inv;
      shift = (bb - bn_m[nn]) * inv + bn_b[nn];
    }
#pragma unroll
    for (int r = 0; r < 8; ++r) {
      int mm   = mb + kh * 8 + r;         // C/D layout: m = kh*8 + vgpr, n = lane&15
      int bidx = mm / 36;
      int p    = mm - bidx * 36;
      size_t idx = ((size_t)bidx * N + nn) * 36 + p;
      float v = acc[r] * scale + shift;
      if (flags & FLAG_RELU) v = fmaxf(v, 0.f);
      if (flags & FLAG_ACC)  v += out[idx];
      out[idx] = v;
    }
  };
#pragma unroll
  for (int j = 0; j < 4; ++j) {
    int nn = n0 + j * 16 + nloc;
    epilogue(acc0[j], m0,      nn);
    epilogue(acc1[j], m0 + 16, nn);
  }
}

// --- attention: per sample, alpha = softmax(q . m_flat_d), c = relu(alpha . m_flat)
__global__ __launch_bounds__(256) void attention_kernel(
    float* __restrict__ c_out, const float* __restrict__ q,
    const float* __restrict__ mflat) {
  __shared__ float qs[36];
  __shared__ float lg[336];
  __shared__ float red[256];
  const int b = blockIdx.x, tid = threadIdx.x;
  if (tid < 36) qs[tid] = q[(size_t)b * 36 + tid];
  __syncthreads();
  const float* mb = mflat + (size_t)b * 336 * 36;
  for (int d = tid; d < 336; d += 256) {
    float s = 0.f;
    for (int j = 0; j < 36; ++j) s += qs[j] * mb[(size_t)d * 36 + j];
    lg[d] = s;
  }
  __syncthreads();
  float mx = -1e30f;
  for (int d = tid; d < 336; d += 256) mx = fmaxf(mx, lg[d]);
  red[tid] = mx; __syncthreads();
  for (int s = 128; s > 0; s >>= 1) {
    if (tid < s) red[tid] = fmaxf(red[tid], red[tid + s]);
    __syncthreads();
  }
  mx = red[0]; __syncthreads();
  float sm = 0.f;
  for (int d = tid; d < 336; d += 256) { float e = __expf(lg[d] - mx); lg[d] = e; sm += e; }
  red[tid] = sm; __syncthreads();
  for (int s = 128; s > 0; s >>= 1) {
    if (tid < s) red[tid] += red[tid + s];
    __syncthreads();
  }
  float inv = 1.f / red[0];
  if (tid < 36) {
    float acc = 0.f;
    for (int d = 0; d < 336; ++d) acc += lg[d] * mb[(size_t)d * 36 + tid];
    c_out[(size_t)b * 36 + tid] = fmaxf(acc * inv, 0.f);   // torch relu_ on c
  }
}

// --- GRU gating: h_new = (1-z)*tanh(an + r*hn) + z*h
__global__ void gru_final_kernel(float* __restrict__ out,
                                 const float* __restrict__ bR, const float* __restrict__ bZ,
                                 const float* __restrict__ bAN, const float* __restrict__ bHN,
                                 const float* __restrict__ h, long total) {
  long i = (long)blockIdx.x * blockDim.x + threadIdx.x;
  if (i >= total) return;
  float r = 1.f / (1.f + __expf(-bR[i]));
  float z = 1.f / (1.f + __expf(-bZ[i]));
  float n = tanhf(bAN[i] + r * bHN[i]);
  out[i] = (1.f - z) * n + z * h[i];
}

// ---------------------------------------------------------------------------
extern "C" void kernel_launch(void* const* d_in, const int* in_sizes, int n_in,
                              void* d_out, int out_size, void* d_ws, size_t ws_size,
                              hipStream_t stream) {
  (void)in_sizes; (void)n_in; (void)out_size; (void)ws_size;
  // Input order: jax.tree_util.tree_leaves(setup_inputs()) -> sorted dict keys:
  // 'm', 'params'{gru{hn,hn_b,hr,hr_b,hz,hz_b,in,in_b,ir,ir_b,iz,iz_b}, out_b,
  //  out_w, q_b, q_w, reader[{b,bn{b,g,m,v},w}]x3, wi_b, wi_w, writer[...]x3}, 's'
  const float* m_in  = (const float*)d_in[0];
  const float* g_hn  = (const float*)d_in[1];  const float* g_hn_b = (const float*)d_in[2];
  const float* g_hr  = (const float*)d_in[3];  const float* g_hr_b = (const float*)d_in[4];
  const float* g_hz  = (const float*)d_in[5];  const float* g_hz_b = (const float*)d_in[6];
  const float* g_in  = (const float*)d_in[7];  const float* g_in_b = (const float*)d_in[8];
  const float* g_ir  = (const float*)d_in[9];  const float* g_ir_b = (const float*)d_in[10];
  const float* g_iz  = (const float*)d_in[11]; const float* g_iz_b = (const float*)d_in[12];
  const float* out_b = (const float*)d_in[13]; const float* out_w  = (const float*)d_in[14];
  const float* q_b   = (const float*)d_in[15]; const float* q_w    = (const float*)d_in[16];
  const float* r0_b  = (const float*)d_in[17];
  const float* r0_bnb = (const float*)d_in[18]; const float* r0_bng = (const float*)d_in[19];
  const float* r0_bnm = (const float*)d_in[20]; const float* r0_bnv = (const float*)d_in[21];
  const float* r0_w  = (const float*)d_in[22];
  const float* r1_b  = (const float*)d_in[23];
  const float* r1_bnb = (const float*)d_in[24]; const float* r1_bng = (const float*)d_in[25];
  const float* r1_bnm = (const float*)d_in[26]; const float* r1_bnv = (const float*)d_in[27];
  const float* r1_w  = (const float*)d_in[28];
  const float* r2_b  = (const float*)d_in[29];
  const float* r2_bnb = (const float*)d_in[30]; const float* r2_bng = (const float*)d_in[31];
  const float* r2_bnm = (const float*)d_in[32]; const float* r2_bnv = (const float*)d_in[33];
  const float* r2_w  = (const float*)d_in[34];
  const float* wi_b  = (const float*)d_in[35]; const float* wi_w   = (const float*)d_in[36];
  const float* w0_b  = (const float*)d_in[37];
  const float* w0_bnb = (const float*)d_in[38]; const float* w0_bng = (const float*)d_in[39];
  const float* w0_bnm = (const float*)d_in[40]; const float* w0_bnv = (const float*)d_in[41];
  const float* w0_w  = (const float*)d_in[42];
  const float* w1_b  = (const float*)d_in[43];
  const float* w1_bnb = (const float*)d_in[44]; const float* w1_bng = (const float*)d_in[45];
  const float* w1_bnm = (const float*)d_in[46]; const float* w1_bnv = (const float*)d_in[47];
  const float* w1_w  = (const float*)d_in[48];
  const float* w2_b  = (const float*)d_in[49];
  const float* w2_bnb = (const float*)d_in[50]; const float* w2_bng = (const float*)d_in[51];
  const float* w2_bnm = (const float*)d_in[52]; const float* w2_bnv = (const float*)d_in[53];
  const float* w2_w  = (const float*)d_in[54];
  const float* s_ptr = (const float*)d_in[55];

  const size_t M = MTOT;
  float* o_out = (float*)d_out;                       // (1024,32,36)
  float* h_out = o_out + (size_t)1024 * 32 * 36;      // (1024,336,36)

  // ---- workspace bump allocator ----
  unsigned char* ws = (unsigned char*)d_ws;
  size_t off = 0;
  auto alloc = [&](size_t bytes) -> void* {
    void* p = ws + off;
    off += (bytes + 255) & ~(size_t)255;
    return p;
  };
  // bf16 weight buffers (N x Kpad)
  unsigned short* wb_r0  = (unsigned short*)alloc((size_t)448 * 3040 * 2);
  unsigned short* wb_r1  = (unsigned short*)alloc((size_t)112 * 4032 * 2);
  unsigned short* wb_r2  = (unsigned short*)alloc((size_t)1   * 1024 * 2);
  unsigned short* wb_q   = (unsigned short*)alloc((size_t)1   * 320  * 2);
  unsigned short* wb_wi  = (unsigned short*)alloc((size_t)1   * 288  * 2);
  unsigned short* wb_w0  = (unsigned short*)alloc((size_t)112 * 32   * 2);
  unsigned short* wb_w1  = (unsigned short*)alloc((size_t)448 * 1024 * 2);
  unsigned short* wb_w2  = (unsigned short*)alloc((size_t)336 * 4032 * 2);
  unsigned short* wb_out = (unsigned short*)alloc((size_t)32  * 64   * 2);
  unsigned short* wb_gir = (unsigned short*)alloc((size_t)336 * 3040 * 2);
  unsigned short* wb_giz = (unsigned short*)alloc((size_t)336 * 3040 * 2);
  unsigned short* wb_gin = (unsigned short*)alloc((size_t)336 * 3040 * 2);
  unsigned short* wb_ghr = (unsigned short*)alloc((size_t)336 * 3040 * 2);
  unsigned short* wb_ghz = (unsigned short*)alloc((size_t)336 * 3040 * 2);
  unsigned short* wb_ghn = (unsigned short*)alloc((size_t)336 * 3040 * 2);
  // im2col buffers (M x Kpad, bf16)
  unsigned short* colH = (unsigned short*)alloc(M * 3040 * 2);  // im2col(h), reused by GRU
  unsigned short* colA = (unsigned short*)alloc(M * 4032 * 2);  // scratch, reused per layer
  // fp32 activations
  float* x1    = (float*)alloc(M * 448 * 4);   // reader L1 out; later reused as w2buf
  float* x2    = (float*)alloc(M * 112 * 4);   // reader L2 out; later reused as w1buf
  float* rbuf  = (float*)alloc(M * 4);
  float* qbuf  = (float*)alloc(M * 4);
  float* cbuf  = (float*)alloc(M * 4);
  float* sinb  = (float*)alloc(M * 4);
  float* cat33 = (float*)alloc(M * 33 * 4);
  float* cat3  = (float*)alloc(M * 3  * 4);
  float* cat34 = (float*)alloc(M * 34 * 4);
  float* wact  = (float*)alloc(M * 336 * 4);
  float* bufR  = (float*)alloc(M * 336 * 4);
  float* bufZ  = (float*)alloc(M * 336 * 4);
  float* bufAN = (float*)alloc(M * 336 * 4);
  float* bufHN = (float*)alloc(M * 336 * 4);
  float* w1buf = x2;   // alias (x2 dead after reader L3 im2col)
  float* w2buf = x1;   // alias (x1 dead after reader L2 im2col)

  auto ewg = [](long total) { return dim3((unsigned)((total + 255) / 256)); };
  auto prep = [&](unsigned short* dst, const float* src, int N, int Cin, int Kpad, int mode) {
    long t = (long)N * Kpad;
    prep_w_kernel<<<ewg(t), 256, 0, stream>>>(dst, src, N, Cin, Kpad, mode, t);
  };
  auto im2col = [&](unsigned short* dst, const float* src, int C, int Kpad, int ks3) {
    long t = (long)M * Kpad;
    im2col_kernel<<<ewg(t), 256, 0, stream>>>(dst, src, C, Kpad, ks3, t);
  };
  auto catcp = [&](float* dst, const float* src, int Ctot, int choff, int C) {
    long t = (long)1024 * C * 36;
    copy_ch_kernel<<<ewg(t), 256, 0, stream>>>(dst, src, Ctot, choff, C, t);
  };
  auto gemm = [&](float* out, const unsigned short* A, const unsigned short* Bw,
                  int N, int Kpad, const float* bias, const float* bng, const float* bnb,
                  const float* bnm, const float* bnv, int flags) {
    dim3 grid((N + 127) / 128, MBLOCKS);
    gemm_wmma_kernel<<<grid, 256, 0, stream>>>(out, A, Bw, N, Kpad, bias,
                                               bng, bnb, bnm, bnv, flags);
  };

  // ---- weight prep (bf16, K padded to mult of 32) ----
  prep(wb_r0, r0_w, 448, 336, 3040, 0);
  prep(wb_r1, r1_w, 112, 448, 4032, 0);
  prep(wb_r2, r2_w, 1,   112, 1024, 0);
  prep(wb_q,  q_w,  1,   33,  320,  0);
  prep(wb_wi, wi_w, 1,   32,  288,  0);
  prep(wb_w0, w0_w, 112, 3,   32,   1);   // convT: flip + transpose
  prep(wb_w1, w1_w, 448, 112, 1024, 1);
  prep(wb_w2, w2_w, 336, 448, 4032, 1);
  prep(wb_out, out_w, 32, 34, 64,   2);   // 1x1
  prep(wb_gir, g_ir, 336, 336, 3040, 0);
  prep(wb_giz, g_iz, 336, 336, 3040, 0);
  prep(wb_gin, g_in, 336, 336, 3040, 0);
  prep(wb_ghr, g_hr, 336, 336, 3040, 0);
  prep(wb_ghz, g_hz, 336, 336, 3040, 0);
  prep(wb_ghn, g_hn, 336, 336, 3040, 0);

  // ---- reader: h = merge(m) = m reshaped (B,336,36) since MF==1 ----
  im2col(colH, m_in, 336, 3040, 1);  // kept live: also feeds GRU h-convs
  gemm(x1, colH, wb_r0, 448, 3040, r0_b, r0_bng, r0_bnb, r0_bnm, r0_bnv, FLAG_BN | FLAG_RELU);
  im2col(colA, x1, 448, 4032, 1);
  gemm(x2, colA, wb_r1, 112, 4032, r1_b, r1_bng, r1_bnb, r1_bnm, r1_bnv, FLAG_BN | FLAG_RELU);
  im2col(colA, x2, 112, 1024, 1);
  gemm(rbuf, colA, wb_r2, 1, 1024, r2_b, r2_bng, r2_bnb, r2_bnm, r2_bnv, FLAG_BN | FLAG_RELU);

  // ---- attention: q = relu(conv([s;r])); alpha = softmax; c = relu(alpha.m) ----
  catcp(cat33, s_ptr, 33, 0, 32);
  catcp(cat33, rbuf,  33, 32, 1);
  im2col(colA, cat33, 33, 320, 1);
  gemm(qbuf, colA, wb_q, 1, 320, q_b, nullptr, nullptr, nullptr, nullptr, FLAG_RELU);
  attention_kernel<<<1024, 256, 0, stream>>>(cbuf, qbuf, m_in);

  // ---- write-interface: s_in = relu(conv(s)) ----
  im2col(colA, s_ptr, 32, 288, 1);
  gemm(sinb, colA, wb_wi, 1, 288, wi_b, nullptr, nullptr, nullptr, nullptr, FLAG_RELU);

  // ---- writer (convT stack) ----
  catcp(cat3, cbuf, 3, 0, 1);
  catcp(cat3, rbuf, 3, 1, 1);
  catcp(cat3, sinb, 3, 2, 1);
  im2col(colA, cat3, 3, 32, 1);
  gemm(w1buf, colA, wb_w0, 112, 32, w0_b, w0_bng, w0_bnb, w0_bnm, w0_bnv, FLAG_BN | FLAG_RELU);
  im2col(colA, w1buf, 112, 1024, 1);
  gemm(w2buf, colA, wb_w1, 448, 1024, w1_b, w1_bng, w1_bnb, w1_bnm, w1_bnv, FLAG_BN | FLAG_RELU);
  im2col(colA, w2buf, 448, 4032, 1);
  gemm(wact, colA, wb_w2, 336, 4032, w2_b, w2_bng, w2_bnb, w2_bnm, w2_bnv, FLAG_BN | FLAG_RELU);

  // ---- output interface: o = relu(1x1 conv over [c,r,s]) -> directly into d_out ----
  catcp(cat34, cbuf,  34, 0, 1);
  catcp(cat34, rbuf,  34, 1, 1);
  catcp(cat34, s_ptr, 34, 2, 32);
  im2col(colA, cat34, 34, 64, 0);   // 1x1 -> layout transform only
  gemm(o_out, colA, wb_out, 32, 64, out_b, nullptr, nullptr, nullptr, nullptr, FLAG_RELU);

  // ---- ConvGRU: gates from conv(w,*) then accumulate conv(h,*) ----
  im2col(colA, wact, 336, 3040, 1);
  gemm(bufR,  colA, wb_gir, 336, 3040, g_ir_b, nullptr, nullptr, nullptr, nullptr, 0);
  gemm(bufZ,  colA, wb_giz, 336, 3040, g_iz_b, nullptr, nullptr, nullptr, nullptr, 0);
  gemm(bufAN, colA, wb_gin, 336, 3040, g_in_b, nullptr, nullptr, nullptr, nullptr, 0);
  gemm(bufR,  colH, wb_ghr, 336, 3040, g_hr_b, nullptr, nullptr, nullptr, nullptr, FLAG_ACC);
  gemm(bufZ,  colH, wb_ghz, 336, 3040, g_hz_b, nullptr, nullptr, nullptr, nullptr, FLAG_ACC);
  gemm(bufHN, colH, wb_ghn, 336, 3040, g_hn_b, nullptr, nullptr, nullptr, nullptr, 0);

  long th = (long)M * 336;
  gru_final_kernel<<<ewg(th), 256, 0, stream>>>(h_out, bufR, bufZ, bufAN, bufHN, m_in, th);
}
